// BiLSTM_CRF_BERT_47845935677741
// MI455X (gfx1250) — matrix operand order repaired and checked by reading, not compiled, not run on hardware
//
#include <hip/hip_runtime.h>
#include <cstddef>

// ---------------------------------------------------------------------------
// Problem constants
// ---------------------------------------------------------------------------
#define BB   64      // batch
#define SS   512     // seq len
#define EE   768     // embed dim
#define HDD  384     // per-direction hidden
#define G4   1536    // 4*HD
#define TT   20      // tags
#define START_TAG 17
#define STOP_TAG  18
#define NEGV (-10000.0f)

typedef __attribute__((ext_vector_type(16))) __bf16 v16bf;
typedef __attribute__((ext_vector_type(8)))  float  v8f;

// ---------------------------------------------------------------------------
// WMMA fragment helpers (CDNA5 ISA 7.12.2 layouts, wave32)
// ---------------------------------------------------------------------------
__device__ __forceinline__ v16bf frag_a(const __bf16* lds, int row0, int k0, int ld) {
    int lane  = threadIdx.x & 31;
    int row   = row0 + (lane & 15);
    int khalf = (lane >> 4) & 1;
    v16bf f;
#pragma unroll
    for (int e = 0; e < 16; ++e) {
        int v = e >> 1, pos = e & 1;
        int k = k0 + ((v & 3) * 2 + pos) + ((v >= 4) ? 16 : 0) + khalf * 8;
        f[e] = lds[row * ld + k];
    }
    return f;
}

// B stored N-major in LDS: [n][k]
__device__ __forceinline__ v16bf frag_b(const __bf16* lds, int n0, int k0, int ld) {
    int lane  = threadIdx.x & 31;
    int n     = n0 + (lane & 15);
    int khalf = (lane >> 4) & 1;
    v16bf f;
#pragma unroll
    for (int e = 0; e < 16; ++e) {
        int k = k0 + khalf * 8 + (e & 7) + ((e >> 3) * 16);
        f[e] = lds[n * ld + k];
    }
    return f;
}

__device__ __forceinline__ v8f wmma_bf16(v16bf a, v16bf b, v8f c) {
    return __builtin_amdgcn_wmma_f32_16x16x32_bf16(false, a, false, b,
                                                   (short)0, c, false, false);
}

// Async global->LDS copy of one 16-byte packet (GLOBAL_LOAD_ASYNC_TO_LDS_B128,
// GV mode; tracked by ASYNCcnt). ldsa = LDS byte address (flat addr low bits),
// ga = 64-bit global byte address.
__device__ __forceinline__ void async_g2l_b128(unsigned ldsa, unsigned long long ga) {
#if defined(__AMDGCN__)
    asm volatile("global_load_async_to_lds_b128 %0, %1, off"
                 :: "v"(ldsa), "v"(ga) : "memory");
#endif
}

__device__ __forceinline__ void wait_asynccnt0() {
#if defined(__AMDGCN__)
    asm volatile("s_wait_asynccnt 0" ::: "memory");
#endif
}

// ---------------------------------------------------------------------------
// Kernel 0: init -- zero step counters, seed h0 slots (bf16)
// ---------------------------------------------------------------------------
__global__ void k_init(const float* __restrict__ h0,
                       __bf16* hallf, __bf16* hallb, unsigned* counters) {
    int idx = blockIdx.x * blockDim.x + threadIdx.x;
    if (idx < 2 * SS) counters[idx] = 0u;
    if (idx < 2 * BB * HDD) {
        int d = idx / (BB * HDD);
        int r = idx % (BB * HDD);
        __bf16 v = (__bf16)h0[idx];
        if (d == 0) hallf[r] = v;                              // fwd slot 0
        else        hallb[(size_t)SS * BB * HDD + r] = v;      // bwd slot S
    }
}

// ---------------------------------------------------------------------------
// Kernel 1: Xproj = emb @ W_ih^T + b_ih + b_hh   (both directions, bf16 out)
// Block: 256 thr (8 waves), 128x128 tile; waves 4(M)x2(N), each 2x4 subtiles
// ---------------------------------------------------------------------------
__global__ __launch_bounds__(256)
void k_xproj(const float* __restrict__ emb,
             const float* __restrict__ Wf, const float* __restrict__ Wb,
             const float* __restrict__ bihf, const float* __restrict__ bhhf,
             const float* __restrict__ bihb, const float* __restrict__ bhhb,
             __bf16* xprojf, __bf16* xprojb) {
    const int dir = blockIdx.z;
    const float* W   = dir ? Wb   : Wf;
    const float* bih = dir ? bihb : bihf;
    const float* bhh = dir ? bhhb : bhhf;
    __bf16* out      = dir ? xprojb : xprojf;

    __shared__ __bf16 sA[128 * 40];  // [row][k], ld=40
    __shared__ __bf16 sB[128 * 40];  // [n][k],   ld=40

    const int tid  = threadIdx.x;
    const int wave = tid >> 5;
    const int lane = tid & 31;
    const int wm = wave & 3;   // 2 Mtiles at wm*32
    const int wn = wave >> 2;  // 4 Ntiles at wn*64
    const int m0 = blockIdx.x * 128;
    const int n0 = blockIdx.y * 128;

    v8f acc[2][4];
#pragma unroll
    for (int i = 0; i < 2; ++i)
#pragma unroll
        for (int j = 0; j < 4; ++j) acc[i][j] = {};

    for (int k0 = 0; k0 < EE; k0 += 32) {
        // stage A (embedding rows, f32->bf16). A row m = s*B + b; emb is [B,S,E]
        {
            int r    = tid >> 1;
            int kcb  = (tid & 1) * 16;
            int m    = m0 + r;
            int s    = m >> 6, b = m & 63;
            const float* src = emb + ((size_t)b * SS + s) * EE + k0 + kcb;
#pragma unroll
            for (int j = 0; j < 4; ++j) {
                float4 v = *(const float4*)(src + j * 4);
                sA[r * 40 + kcb + j * 4 + 0] = (__bf16)v.x;
                sA[r * 40 + kcb + j * 4 + 1] = (__bf16)v.y;
                sA[r * 40 + kcb + j * 4 + 2] = (__bf16)v.z;
                sA[r * 40 + kcb + j * 4 + 3] = (__bf16)v.w;
            }
        }
        // stage B (W rows = output cols), W is [G4][E]
        {
            int n   = tid >> 1;
            int kcb = (tid & 1) * 16;
            const float* src = W + (size_t)(n0 + n) * EE + k0 + kcb;
#pragma unroll
            for (int j = 0; j < 4; ++j) {
                float4 v = *(const float4*)(src + j * 4);
                sB[n * 40 + kcb + j * 4 + 0] = (__bf16)v.x;
                sB[n * 40 + kcb + j * 4 + 1] = (__bf16)v.y;
                sB[n * 40 + kcb + j * 4 + 2] = (__bf16)v.z;
                sB[n * 40 + kcb + j * 4 + 3] = (__bf16)v.w;
            }
        }
        __syncthreads();

        v16bf a0 = frag_a(sA, wm * 32,      0, 40);
        v16bf a1 = frag_a(sA, wm * 32 + 16, 0, 40);
#pragma unroll
        for (int j = 0; j < 4; ++j) {
            v16bf bf = frag_b(sB, wn * 64 + j * 16, 0, 40);
            acc[0][j] = wmma_bf16(a0, bf, acc[0][j]);
            acc[1][j] = wmma_bf16(a1, bf, acc[1][j]);
        }
        __syncthreads();
    }

    // epilogue: + bias, store bf16
    const int khalf = (lane >> 4) & 1;
#pragma unroll
    for (int i = 0; i < 2; ++i)
#pragma unroll
        for (int j = 0; j < 4; ++j) {
            int n  = n0 + wn * 64 + j * 16 + (lane & 15);
            float bias = bih[n] + bhh[n];
#pragma unroll
            for (int v = 0; v < 8; ++v) {
                int m = m0 + wm * 32 + i * 16 + v + 8 * khalf;
                out[(size_t)m * G4 + n] = (__bf16)(acc[i][j][v] + bias);
            }
        }
}

// ---------------------------------------------------------------------------
// Kernel 2: persistent bidirectional LSTM recurrence.
// grid (8 WGs per dir, 2 dirs). Each WG owns a 48-unit slice of HD and its
// W_hh columns for all 4 gates, resident in LDS (bf16) for all 512 steps
// (CDNA5: 320KB LDS per WGP makes this possible). h is exchanged through
// global (L2) each step via ASYNC global->LDS loads; cross-WG sync = atomic
// release/acquire barrier + s_cluster_barrier.
// ---------------------------------------------------------------------------
#define LSTM_SMEM (150528 + 49152 + 12288 + 51200)   // sW + sH + sC + sG

__global__ __launch_bounds__(256)
void k_lstm(const float* __restrict__ c0,
            const float* __restrict__ Whhf, const float* __restrict__ Whhb,
            const __bf16* __restrict__ xprojf, const __bf16* __restrict__ xprojb,
            __bf16* hallf, __bf16* hallb, unsigned* counters) {
    const int dir = blockIdx.y;
    const int wg  = blockIdx.x;         // 0..7
    const int n0  = wg * 48;
    const int tid = threadIdx.x;
    const int wave = tid >> 5;
    const int lane = tid & 31;

    const float*  Whh = dir ? Whhb : Whhf;
    const __bf16* xp  = dir ? xprojb : xprojf;
    __bf16*       hall = dir ? hallb : hallf;
    unsigned*     ctr0 = counters + dir * SS;

    extern __shared__ char smem[];
    __bf16* sW = (__bf16*)smem;                         // [192][392] B: [col][k]
    __bf16* sH = (__bf16*)(smem + 150528);              // [64][384]  A: [b][k]
    float*  sC = (float*)(smem + 150528 + 49152);       // [64*48] cell state
    float*  sG = (float*)(smem + 150528 + 49152 + 12288); // [64][200] gates f32

    // LDS byte address of sH (flat address low 32 bits, ISA 10.2) for async DMA
    const unsigned sH_lds = (unsigned)(unsigned long long)(void*)sH;

    // one-time: stage W_hh slice (gate-major columns) into LDS as bf16
    for (int idx = tid; idx < 192 * HDD; idx += 256) {
        int col = idx / HDD, k = idx % HDD;
        int g = col / 48, nn = col % 48;
        int n = g * HDD + n0 + nn;
        sW[col * 392 + k] = (__bf16)Whh[(size_t)n * HDD + k];
    }
    // one-time: load cell-state slice
    for (int idx = tid; idx < BB * 48; idx += 256) {
        int b = idx / 48, nn = idx % 48;
        sC[idx] = c0[(size_t)dir * BB * HDD + b * HDD + n0 + nn];
    }
    __syncthreads();

    const int mt = wave & 3;   // Mtile (16 batch rows)
    const int cg = wave >> 2;  // col-tile group: 6 tiles each
    const int khalf = (lane >> 4) & 1;

    for (int t = 0; t < SS; ++t) {
        const int s         = dir ? (SS - 1 - t) : t;
        const int slot_prev = dir ? (s + 1) : s;
        const int slot_new  = dir ? s : (s + 1);

        // stage h_prev with async global->LDS b128 DMA (49152 B = 3072 packets)
        {
            unsigned long long src =
                (unsigned long long)(const void*)(hall + (size_t)slot_prev * BB * HDD);
#pragma unroll
            for (int it = 0; it < 12; ++it) {
                unsigned byteoff = (unsigned)(tid + it * 256) * 16u;
                async_g2l_b128(sH_lds + byteoff, src + byteoff);
            }
            wait_asynccnt0();
        }
        __syncthreads();

        v8f acc[6];
#pragma unroll
        for (int j = 0; j < 6; ++j) acc[j] = {};

#pragma unroll 4
        for (int k0 = 0; k0 < HDD; k0 += 32) {
            v16bf a = frag_a(sH, mt * 16, k0, HDD);
#pragma unroll
            for (int j = 0; j < 6; ++j) {
                v16bf bf = frag_b(sW, (cg * 6 + j) * 16, k0, 392);
                acc[j] = wmma_bf16(a, bf, acc[j]);
            }
        }

        // add input projection, spill gates (f32) to LDS for recombination
        const size_t xrow = (size_t)s * BB;
#pragma unroll
        for (int j = 0; j < 6; ++j) {
            int col = (cg * 6 + j) * 16 + (lane & 15);     // 0..191 gate-major
            int g = col / 48, nn = col % 48;
            int gc = g * HDD + n0 + nn;                     // global gate column
#pragma unroll
            for (int v = 0; v < 8; ++v) {
                int b = mt * 16 + v + 8 * khalf;
                sG[b * 200 + col] = acc[j][v] + (float)xp[(xrow + b) * G4 + gc];
            }
        }
        __syncthreads();

        // elementwise LSTM cell: gates order i,f,g,o
        for (int idx = tid; idx < BB * 48; idx += 256) {
            int b = idx / 48, nn = idx % 48;
            float gi = sG[b * 200 +       nn];
            float gf = sG[b * 200 +  48 + nn];
            float gg = sG[b * 200 +  96 + nn];
            float go = sG[b * 200 + 144 + nn];
            gi = 1.0f / (1.0f + __expf(-gi));
            gf = 1.0f / (1.0f + __expf(-gf));
            go = 1.0f / (1.0f + __expf(-go));
            float c = gf * sC[idx] + gi * tanhf(gg);
            sC[idx] = c;
            float h = go * tanhf(c);
            hall[(size_t)slot_new * BB * HDD + b * HDD + n0 + nn] = (__bf16)h;
        }

        // release -> inter-WG barrier (same direction) -> acquire
        __threadfence();
        __syncthreads();
        if (tid == 0) {
            unsigned* c = ctr0 + t;
            atomicAdd(c, 1u);
            while (atomicAdd(c, 0u) < 8u) __builtin_amdgcn_s_sleep(2);
        }
        __builtin_amdgcn_s_cluster_barrier();   // real barrier under cluster dispatch
        __syncthreads();
        __threadfence();

        if (t + 1 < SS) {   // warm L2/L0 for next step's Xproj slice
            int sn = dir ? (s - 1) : (s + 1);
            __builtin_prefetch(&xp[((size_t)sn * BB) * G4 + n0], 0, 0);
        }
    }
}

// ---------------------------------------------------------------------------
// Kernel 3: feats[b,s,t] = lstm_out[s,b,:] . W_out[t,:] + b_out[t]
// Block: 128 thr (4 waves), 64 rows x 32 cols (tags padded to 32)
// ---------------------------------------------------------------------------
__global__ __launch_bounds__(128)
void k_feats(const __bf16* __restrict__ hallf, const __bf16* __restrict__ hallb,
             const float* __restrict__ Wout, const float* __restrict__ bout,
             float* __restrict__ feats) {
    __shared__ __bf16 sW[32 * 776];   // [t][k], zero padded t>=20
    __shared__ __bf16 sA[64 * 40];

    const int tid  = threadIdx.x;
    const int wave = tid >> 5;
    const int lane = tid & 31;
    const int m0   = blockIdx.x * 64;

    for (int idx = tid; idx < 32 * EE; idx += 128) {
        int tt = idx / EE, k = idx % EE;
        sW[tt * 776 + k] = (tt < TT) ? (__bf16)Wout[(size_t)tt * EE + k] : (__bf16)0.0f;
    }
    __syncthreads();

    v8f acc0 = {}, acc1 = {};
    for (int k0 = 0; k0 < 2 * HDD; k0 += 32) {
        // stage A from the two direction trajectories (already bf16)
        for (int idx = tid; idx < 64 * 32; idx += 128) {
            int r = idx >> 5, kk = idx & 31;
            int m = m0 + r, s = m >> 6, b = m & 63;
            int h = k0 + kk;
            __bf16 v = (h < HDD)
                ? hallf[((size_t)(s + 1) * BB + b) * HDD + h]
                : hallb[((size_t)s * BB + b) * HDD + (h - HDD)];
            sA[r * 40 + kk] = v;
        }
        __syncthreads();
        v16bf a  = frag_a(sA, wave * 16, 0, 40);
        v16bf b0 = frag_b(sW,  0, k0, 776);
        v16bf b1 = frag_b(sW, 16, k0, 776);
        acc0 = wmma_bf16(a, b0, acc0);
        acc1 = wmma_bf16(a, b1, acc1);
        __syncthreads();
    }

    const int khalf = (lane >> 4) & 1;
#pragma unroll
    for (int j = 0; j < 2; ++j) {
        const v8f& acc = j ? acc1 : acc0;
        int tg = j * 16 + (lane & 15);
        if (tg < TT) {
            float bo = bout[tg];
#pragma unroll
            for (int v = 0; v < 8; ++v) {
                int m = m0 + wave * 16 + v + 8 * khalf;
                int s = m >> 6, b = m & 63;
                feats[((size_t)b * SS + s) * TT + tg] = acc[v] + bo;
            }
        }
    }
}

// ---------------------------------------------------------------------------
// Kernel 4: CRF forward-alg + Viterbi + backtrace. One wave per batch elem.
// ---------------------------------------------------------------------------
__global__ __launch_bounds__(32)
void k_crf(const float* __restrict__ feats, const float* __restrict__ trans,
           float* __restrict__ out) {
    const int b = blockIdx.x;
    const int i = threadIdx.x;
    __shared__ float st[TT * TT];
    __shared__ unsigned char bps[SS * TT];

    for (int idx = i; idx < TT * TT; idx += 32) st[idx] = trans[idx];
    __syncthreads();

    const int ir = (i < TT) ? i : 0;
    float fv = (i == START_TAG) ? 0.0f : NEGV;   // forward-alg score
    float vv = fv;                               // viterbi score

    for (int s = 0; s < SS; ++s) {
        float feat = (i < TT) ? feats[((size_t)b * SS + s) * TT + i] : NEGV;
        float m = -3.4e38f, vbest = -3.4e38f;
        int bj = 0;
#pragma unroll 4
        for (int j = 0; j < TT; ++j) {
            float tij = st[ir * TT + j];
            float fj = __shfl(fv, j, 32) + tij;
            float vj = __shfl(vv, j, 32) + tij;
            m = fmaxf(m, fj);
            if (vj > vbest) { vbest = vj; bj = j; }
        }
        float ssum = 0.0f;
#pragma unroll 4
        for (int j = 0; j < TT; ++j)
            ssum += __expf(__shfl(fv, j, 32) + st[ir * TT + j] - m);
        fv = feat + m + __logf(ssum);
        vv = feat + vbest;
        if (i < TT) bps[s * TT + i] = (unsigned char)bj;
    }
    __syncthreads();

    float term  = (i < TT) ? (vv + st[STOP_TAG * TT + ir]) : -3.4e38f;
    float aterm = (i < TT) ? (fv + st[STOP_TAG * TT + ir]) : -3.4e38f;

    float bestv = -3.4e38f, am = -3.4e38f;
    int bestj = 0;
#pragma unroll 4
    for (int j = 0; j < TT; ++j) {
        float tj = __shfl(term, j, 32);
        if (tj > bestv) { bestv = tj; bestj = j; }
        am = fmaxf(am, __shfl(aterm, j, 32));
    }
    float asum = 0.0f;
#pragma unroll 4
    for (int j = 0; j < TT; ++j) asum += __expf(__shfl(aterm, j, 32) - am);

    if (i == 0) {
        out[b] = bestv;                               // path_score [B,1]
        out[BB + BB * SS + b] = am + __logf(asum);    // alpha [B,1]
        int cur = bestj;
        out[BB + b * SS + (SS - 1)] = (float)cur;     // best_path [B,S]
        for (int s = SS - 2; s >= 0; --s) {
            cur = bps[(s + 1) * TT + cur];
            out[BB + b * SS + s] = (float)cur;
        }
    }
}

// ---------------------------------------------------------------------------
// Host launcher
// ---------------------------------------------------------------------------
extern "C" void kernel_launch(void* const* d_in, const int* in_sizes, int n_in,
                              void* d_out, int out_size, void* d_ws, size_t ws_size,
                              hipStream_t stream) {
    (void)in_sizes; (void)n_in; (void)out_size; (void)ws_size;
    const float* emb   = (const float*)d_in[0];
    const float* h0    = (const float*)d_in[1];
    const float* c0    = (const float*)d_in[2];
    const float* Wihf  = (const float*)d_in[3];
    const float* Whhf  = (const float*)d_in[4];
    const float* bihf  = (const float*)d_in[5];
    const float* bhhf  = (const float*)d_in[6];
    const float* Wihb  = (const float*)d_in[7];
    const float* Whhb  = (const float*)d_in[8];
    const float* bihb  = (const float*)d_in[9];
    const float* bhhb  = (const float*)d_in[10];
    const float* Wout  = (const float*)d_in[11];
    const float* bout  = (const float*)d_in[12];
    const float* trans = (const float*)d_in[13];
    float* out = (float*)d_out;

    // workspace layout (bytes)
    char* ws = (char*)d_ws;
    const size_t XPROJ_SZ = (size_t)SS * BB * G4 * sizeof(__bf16);        // 100.7 MB
    const size_t HALL_SZ  = (size_t)(SS + 1) * BB * HDD * sizeof(__bf16); // 25.2 MB
    __bf16* xprojf = (__bf16*)(ws);
    __bf16* xprojb = (__bf16*)(ws + XPROJ_SZ);
    __bf16* hallf  = (__bf16*)(ws + 2 * XPROJ_SZ);
    __bf16* hallb  = (__bf16*)(ws + 2 * XPROJ_SZ + HALL_SZ);
    float*  feats  = (float*)(ws + 2 * XPROJ_SZ + 2 * HALL_SZ);
    unsigned* ctrs = (unsigned*)(ws + 2 * XPROJ_SZ + 2 * HALL_SZ
                                 + (size_t)BB * SS * TT * sizeof(float));

    static int smem_set = 0;   // attribute set is idempotent & not a stream op
    if (!smem_set) {
        hipFuncSetAttribute((const void*)k_lstm,
                            hipFuncAttributeMaxDynamicSharedMemorySize, LSTM_SMEM);
        smem_set = 1;
    }

    k_init<<<192, 256, 0, stream>>>(h0, hallf, hallb, ctrs);

    k_xproj<<<dim3((SS * BB) / 128, G4 / 128, 2), 256, 0, stream>>>(
        emb, Wihf, Wihb, bihf, bhhf, bihb, bhhb, xprojf, xprojb);

    k_lstm<<<dim3(8, 2), 256, LSTM_SMEM, stream>>>(
        c0, Whhf, Whhb, xprojf, xprojb, hallf, hallb, ctrs);

    k_feats<<<(SS * BB) / 64, 128, 0, stream>>>(hallf, hallb, Wout, bout, feats);

    k_crf<<<BB, 32, 0, stream>>>(feats, trans, out);
}
